// UNetMidBlockCausal3D_11012296147461
// MI455X (gfx1250) — compile-verified, matrix operand order checked by reading, not run on hardware
//
#include <hip/hip_runtime.h>
#include <hip/hip_bf16.h>

// ---------------------------------------------------------------------------
// Problem constants
// ---------------------------------------------------------------------------
#define Cc   512
#define Tt   9
#define Hh   32
#define Ww   32
#define Ll   (Tt * Hh * Ww)          // 9216
#define SZ   ((size_t)Cc * Ll)       // 4718592
#define CONVK (Cc * 27)              // 13824
#define QSCALE 0.044194173824159216f // 1/sqrt(512)

typedef __attribute__((ext_vector_type(16))) __bf16 v16bf;
typedef __attribute__((ext_vector_type(8)))  float  v8f;
typedef __attribute__((ext_vector_type(4)))  unsigned int v4u;
typedef __attribute__((ext_vector_type(8)))  int v8i_t;
typedef __attribute__((ext_vector_type(4)))  int v4i_t;

#if defined(__has_builtin)
#if __has_builtin(__builtin_amdgcn_tensor_load_to_lds) && \
    __has_builtin(__builtin_amdgcn_s_wait_tensorcnt)
#define HAVE_TDM 1
#endif
#endif
#ifndef HAVE_TDM
#define HAVE_TDM 0
#endif

union Frag16 { v16bf v; unsigned int u[8]; };

__device__ inline unsigned short f2bf(float f) {
    unsigned int u = __float_as_uint(f);
    u += 0x7FFFu + ((u >> 16) & 1u);          // round-to-nearest-even
    return (unsigned short)(u >> 16);
}

__device__ inline v8f wmma_bf16(v16bf a, v16bf b, v8f c) {
    return __builtin_amdgcn_wmma_f32_16x16x32_bf16(false, a, false, b,
                                                   (short)0, c, false, false);
}

// Load a 16x32 bf16 A/B fragment from an LDS tile stored [row][k] (row stride
// ldk shorts).  CDNA5 16-bit matrix layout: lanes 0-15 hold K=0..7 (v0-3) and
// K=16..23 (v4-7); lanes 16-31 hold K=8..15 and K=24..31.
__device__ inline v16bf load_frag_ab(const unsigned short* lds, int row0,
                                     int ldk, int k0, int lane) {
    const int m    = lane & 15;
    const int half = lane >> 4;
    const unsigned short* rp = lds + (size_t)(row0 + m) * ldk + k0 + half * 8;
    Frag16 f;
#pragma unroll
    for (int j = 0; j < 4; ++j) {
        f.u[j]     = *(const unsigned int*)(rp + 2 * j);
        f.u[4 + j] = *(const unsigned int*)(rp + 16 + 2 * j);
    }
    return f.v;
}

// C/D 16x16 f32 fragment <-> LDS [row][n] (row stride ldn floats).
__device__ inline v8f load_frag_c(const float* lds, int row0, int ldn, int n0,
                                  int lane) {
    const int n    = lane & 15;
    const int half = lane >> 4;
    v8f c;
#pragma unroll
    for (int r = 0; r < 8; ++r)
        c[r] = lds[(size_t)(row0 + r + half * 8) * ldn + n0 + n];
    return c;
}

__device__ inline void store_frag_c(float* lds, int row0, int ldn, int n0,
                                    int lane, v8f c) {
    const int n    = lane & 15;
    const int half = lane >> 4;
#pragma unroll
    for (int r = 0; r < 8; ++r)
        lds[(size_t)(row0 + r + half * 8) * ldn + n0 + n] = c[r];
}

// ---------------------------------------------------------------------------
// fp32 -> bf16 conversion (plain, for projection weights / staging)
// ---------------------------------------------------------------------------
__global__ void f32_to_bf16_kernel(const float* __restrict__ in,
                                   unsigned short* __restrict__ out, int n) {
    for (int i = blockIdx.x * blockDim.x + threadIdx.x; i < n;
         i += gridDim.x * blockDim.x)
        out[i] = f2bf(in[i]);
}

// Conv weight convert + K-permute to spatial-major order:
//   out[cout][r*512 + cin] = bf16(in[cout][cin*27 + r]),  r = dz*9+dy*3+dx
// This makes every 32-wide K block of the implicit GEMM have a uniform tap r,
// so the im2col gather needs only SALU-uniform index math per block.
__global__ void convw_permute_kernel(const float* __restrict__ in,
                                     unsigned short* __restrict__ out) {
    const int n = Cc * CONVK;
    for (int idx = blockIdx.x * blockDim.x + threadIdx.x; idx < n;
         idx += gridDim.x * blockDim.x) {
        int cout = idx / CONVK;
        int k    = idx - cout * CONVK;     // k = cin*27 + r
        int cin  = k / 27;
        int r    = k - cin * 27;
        out[(size_t)cout * CONVK + r * Cc + cin] = f2bf(in[idx]);
    }
}

// ---------------------------------------------------------------------------
// GroupNorm (32 groups of 16 channels) + optional SiLU.
// ---------------------------------------------------------------------------
__global__ __launch_bounds__(256)
void gn_kernel(const float* __restrict__ x, const float* __restrict__ gamma,
               const float* __restrict__ beta, float* __restrict__ outf,
               unsigned short* __restrict__ outb, int silu, int transpose) {
    const int g    = blockIdx.x;
    const int tid  = threadIdx.x;
    const int base = g * 16 * Ll;
    const int N    = 16 * Ll;

    float s = 0.f, s2 = 0.f;
    for (int i = tid; i < N; i += 256) {
        float v = x[base + i];
        s += v;
        s2 += v * v;
    }
    __shared__ float red[256], red2[256];
    __shared__ float smean, srstd;
    red[tid] = s;
    red2[tid] = s2;
    __syncthreads();
    for (int o = 128; o > 0; o >>= 1) {
        if (tid < o) {
            red[tid] += red[tid + o];
            red2[tid] += red2[tid + o];
        }
        __syncthreads();
    }
    if (tid == 0) {
        float mean = red[0] / (float)N;
        float var  = red2[0] / (float)N - mean * mean;
        smean = mean;
        srstd = rsqrtf(var + 1e-6f);
    }
    __syncthreads();
    const float mean = smean, rstd = srstd;

    for (int i = tid; i < N; i += 256) {
        int cl = i / Ll;
        int p  = i - cl * Ll;
        int c  = g * 16 + cl;
        float v = (x[base + i] - mean) * rstd * gamma[c] + beta[c];
        if (silu) v = v / (1.f + __expf(-v));
        if (outf) outf[base + i] = v;
        if (outb) {
            if (transpose) outb[(size_t)p * Cc + c] = f2bf(v);
            else           outb[base + i] = f2bf(v);
        }
    }
}

// ---------------------------------------------------------------------------
// Causal 3x3x3 conv as implicit GEMM with spatial-major K order.
//   out[cout][p] = sum_k' Wperm[cout][k'] * patch[k'][p],  k' = r*512 + cin
// Block tile 64(cout) x 64(pos); register double-buffered tile loads.
// ---------------------------------------------------------------------------
__global__ __launch_bounds__(256)
void conv3d_igemm_kernel(const unsigned short* __restrict__ Wb,   // permuted bf16
                         const unsigned short* __restrict__ Xb,   // [512][9216] bf16
                         const float* __restrict__ bias,
                         const float* __restrict__ residual,      // nullable
                         float* __restrict__ out) {               // [512][9216]
    __shared__ unsigned short As[64 * 40];
    __shared__ unsigned short Bs[64 * 40];

    const int tid  = threadIdx.x;
    const int lane = tid & 31;
    const int wave = tid >> 5;
    const int mi   = wave & 3;
    const int nj   = wave >> 2;

    const int ntile = blockIdx.x * 64;  // position tile
    const int mtile = blockIdx.y * 64;  // cout tile

    v8f acc0 = {0.f, 0.f, 0.f, 0.f, 0.f, 0.f, 0.f, 0.f};
    v8f acc1 = {0.f, 0.f, 0.f, 0.f, 0.f, 0.f, 0.f, 0.f};

    const int ar = tid >> 2;
    const int ak = (tid & 3) * 8;
    const int bn = tid >> 2;
    const int bk = (tid & 3) * 8;
    const int p   = ntile + bn;
    const int z   = p >> 10;
    const int rem = p & 1023;
    const int yy  = rem >> 5;
    const int xx  = rem & 31;

    const unsigned short* Arow = Wb + (size_t)(mtile + ar) * CONVK + ak;

    // Gather 8 bf16 patch elements for K block k0 (tap r uniform per block).
    auto gatherB = [&](int k0, unsigned short* tmp) {
        int r    = k0 >> 9;                 // 0..26, wave-uniform
        int cin0 = (k0 & 511) + bk;
        int dz = r / 9;
        int r2 = r - dz * 9;
        int dy = r2 / 3;
        int dx = r2 - dy * 3;
        int zi = z + dz - 2;  zi = zi < 0 ? 0 : zi;
        int yi = yy + dy - 1; yi = yi < 0 ? 0 : (yi > 31 ? 31 : yi);
        int xi = xx + dx - 1; xi = xi < 0 ? 0 : (xi > 31 ? 31 : xi);
        const unsigned short* src = Xb + (size_t)cin0 * Ll + zi * 1024 + yi * 32 + xi;
#pragma unroll
        for (int i = 0; i < 8; ++i) tmp[i] = src[(size_t)i * Ll];
    };

    // Prime the software pipeline.
    uint4 aReg = *(const uint4*)Arow;
    alignas(16) unsigned short bReg[8];
    gatherB(0, bReg);

    for (int k0 = 0; k0 < CONVK; k0 += 32) {
        *(uint4*)(As + ar * 40 + ak) = aReg;
        *(uint4*)(Bs + bn * 40 + bk) = *(const uint4*)bReg;
        __syncthreads();

        const int kn = k0 + 32;
        if (kn < CONVK) {
            aReg = *(const uint4*)(Arow + kn);
            gatherB(kn, bReg);
            if (kn + 32 < CONVK)
                __builtin_prefetch(Arow + kn + 32, 0, 0);  // global_prefetch_b8
        }

        v16bf a  = load_frag_ab(As, mi * 16, 40, 0, lane);
        v16bf b0 = load_frag_ab(Bs, nj * 32, 40, 0, lane);
        v16bf b1 = load_frag_ab(Bs, nj * 32 + 16, 40, 0, lane);
        acc0 = wmma_bf16(a, b0, acc0);
        acc1 = wmma_bf16(a, b1, acc1);
        __syncthreads();
    }

    const int nc   = lane & 15;
    const int half = lane >> 4;
#pragma unroll
    for (int r = 0; r < 8; ++r) {
        int m = mtile + mi * 16 + r + half * 8;
        int n = ntile + nj * 32 + nc;
        float bv = bias[m];
        size_t i0 = (size_t)m * Ll + n;
        size_t i1 = i0 + 16;
        float v0 = acc0[r] + bv;
        float v1 = acc1[r] + bv;
        if (residual) { v0 += residual[i0]; v1 += residual[i1]; }
        out[i0] = v0;
        out[i1] = v1;
    }
}

// ---------------------------------------------------------------------------
// Projection GEMM: C[p][f] = A[p][c] * W[c][f] + bias[f]
// mode 0: bf16 out [p][512] (scaled)  |  mode 1: f32 transposed + residual
// ---------------------------------------------------------------------------
__global__ __launch_bounds__(256)
void gemm_proj_kernel(const unsigned short* __restrict__ A,
                      const unsigned short* __restrict__ Bw,
                      const float* __restrict__ bias, float scale, int mode,
                      const float* __restrict__ residual,
                      float* __restrict__ outf,
                      unsigned short* __restrict__ outb) {
    __shared__ unsigned short As[64 * 40];
    __shared__ unsigned short Bs[64 * 40];

    const int tid  = threadIdx.x;
    const int lane = tid & 31;
    const int wave = tid >> 5;
    const int mi   = wave & 3;
    const int nj   = wave >> 2;

    const int mtile = blockIdx.x * 64;  // token rows
    const int ntile = blockIdx.y * 64;  // feature cols

    v8f acc0 = {0.f, 0.f, 0.f, 0.f, 0.f, 0.f, 0.f, 0.f};
    v8f acc1 = {0.f, 0.f, 0.f, 0.f, 0.f, 0.f, 0.f, 0.f};

    const int ar  = tid >> 2;
    const int ak  = (tid & 3) * 8;
    const int kr  = tid >> 3;
    const int nc8 = (tid & 7) * 8;

    const unsigned short* Arow = A + (size_t)(mtile + ar) * Cc + ak;
    const unsigned short* Brow = Bw + (size_t)kr * Cc + ntile + nc8;

    uint4 aReg = *(const uint4*)Arow;
    union { uint4 v; unsigned short s[8]; } bReg;
    bReg.v = *(const uint4*)Brow;

    for (int k0 = 0; k0 < Cc; k0 += 32) {
        *(uint4*)(As + ar * 40 + ak) = aReg;
#pragma unroll
        for (int i = 0; i < 8; ++i)
            Bs[(nc8 + i) * 40 + kr] = bReg.s[i];
        __syncthreads();

        const int kn = k0 + 32;
        if (kn < Cc) {
            aReg  = *(const uint4*)(Arow + kn);
            bReg.v = *(const uint4*)(Brow + (size_t)kn * Cc);
        }

        v16bf a  = load_frag_ab(As, mi * 16, 40, 0, lane);
        v16bf b0 = load_frag_ab(Bs, nj * 32, 40, 0, lane);
        v16bf b1 = load_frag_ab(Bs, nj * 32 + 16, 40, 0, lane);
        acc0 = wmma_bf16(a, b0, acc0);
        acc1 = wmma_bf16(a, b1, acc1);
        __syncthreads();
    }

    const int nc   = lane & 15;
    const int half = lane >> 4;
#pragma unroll
    for (int r = 0; r < 8; ++r) {
        int m = mtile + mi * 16 + r + half * 8;
#pragma unroll
        for (int s = 0; s < 2; ++s) {
            int n = ntile + nj * 32 + s * 16 + nc;
            float v = (s == 0 ? acc0[r] : acc1[r]) + bias[n];
            v *= scale;
            if (mode == 0) {
                outb[(size_t)m * Cc + n] = f2bf(v);
            } else {
                size_t idx = (size_t)n * Ll + m;
                outf[idx] = residual[idx] + v;
            }
        }
    }
}

// ---------------------------------------------------------------------------
// TDM helper: async copy of `nelems` 2-byte elements (contiguous) from global
// memory to LDS via the Tensor Data Mover.  Issued per-wave; tracked by
// TENSORcnt.  1D tile: tile_dim0 = nelems, tile_dim1 = 1.
// This toolchain exposes the 6-arg builtin:
//   (uint32x4 g0, int32x8 g1, int32x4 g2, int32x4 g3, int32x8 pad, i32 cpol)
// ---------------------------------------------------------------------------
#if HAVE_TDM
__device__ inline void tdm_copy_2b(unsigned int lds_addr, const void* gaddr,
                                   unsigned int nelems) {
    unsigned long long ga = (unsigned long long)(size_t)gaddr;
    v4u g0;
    g0[0] = 1u;                                   // count = 1 valid descriptor
    g0[1] = lds_addr;                             // LDS byte address
    g0[2] = (unsigned int)(ga & 0xFFFFFFFFu);     // global_addr[31:0]
    g0[3] = (unsigned int)((ga >> 32) & 0x1FFFFFFu) | (2u << 30); // type=2
    v8i_t g1;
    g1[0] = (int)(1u << 16);                      // data_size = 1 (2 bytes)
    g1[1] = (int)((nelems & 0xFFFFu) << 16);      // tensor_dim0[15:0]
    g1[2] = (int)(((nelems >> 16) & 0xFFFFu) | (1u << 16)); // td0 hi | tensor_dim1=1
    g1[3] = (int)((nelems & 0xFFFFu) << 16);      // tile_dim0 (fits 16 bits)
    g1[4] = 1;                                    // tile_dim1 = 1
    g1[5] = (int)nelems;                          // tensor_dim0_stride lo
    g1[6] = 0;
    g1[7] = 0;
    v4i_t z4 = {0, 0, 0, 0};
    v8i_t z8 = {0, 0, 0, 0, 0, 0, 0, 0};
    __builtin_amdgcn_tensor_load_to_lds(g0, g1, z4, z4, z8, 0);
}
#endif

// ---------------------------------------------------------------------------
// Flash attention with frame-causal mask.
// 32 queries per block; 64-key tiles; K tiles double-buffered, loaded by the
// Tensor Data Mover (wave 0) one tile ahead when available.
// ---------------------------------------------------------------------------
__global__ __launch_bounds__(256)
void attention_kernel(const unsigned short* __restrict__ Q,   // [9216][512], pre-scaled
                      const unsigned short* __restrict__ K,   // [9216][512]
                      const unsigned short* __restrict__ V,   // [9216][512]
                      unsigned short* __restrict__ O) {       // [9216][512]
    extern __shared__ unsigned char smem[];
    float* Oacc        = (float*)smem;                 // 32*512 f32
    float* Ss          = Oacc + 32 * 512;              // 32*64  f32
    float* rmax        = Ss + 32 * 64;
    float* rsum        = rmax + 32;
    float* rscale      = rsum + 32;
    unsigned short* Qs = (unsigned short*)(rscale + 32);   // 32*512
    unsigned short* K0 = Qs + 32 * 512;                    // 64*512 (buffer 0)
    unsigned short* K1 = K0 + 64 * 512;                    // 64*512 (buffer 1)
    unsigned short* Vt = K1 + 64 * 512;                    // 512*64
    unsigned short* Ps = Vt + 512 * 64;                    // 32*64

    const int tid  = threadIdx.x;
    const int lane = tid & 31;
    const int wave = tid >> 5;

    const int qbase  = blockIdx.x * 32;
    const int qframe = qbase >> 10;
    const int kTiles = (qframe + 1) * 16;

    // Load Q tile
    {
        int mr = tid >> 3;
        int koff = (tid & 7) * 64;
        const uint4* src = (const uint4*)(Q + (size_t)(qbase + mr) * Cc + koff);
        uint4* dst = (uint4*)(Qs + mr * Cc + koff);
#pragma unroll
        for (int j = 0; j < 8; ++j) dst[j] = src[j];
    }
    for (int i = tid; i < 32 * 512; i += 256) Oacc[i] = 0.f;
    if (tid < 32) { rmax[tid] = -3.0e38f; rsum[tid] = 0.f; }

#if HAVE_TDM
    // TDM prefetch of key tile 0 into buffer 0 (wave 0 only).
    if (wave == 0)
        tdm_copy_2b((unsigned int)(size_t)(void*)K0, K, 64 * Cc);
#endif
    __syncthreads();

    const int smi = wave & 1;
    const int snj = wave >> 1;

    for (int kt = 0; kt < kTiles; ++kt) {
        const int kb = kt * 64;
        unsigned short* Ks = (kt & 1) ? K1 : K0;
#if !HAVE_TDM
        // Fallback: manual synchronous K tile copy.
        {
            int nr = tid >> 2;
            int koff = (tid & 3) * 128;
            const uint4* src = (const uint4*)(K + (size_t)(kb + nr) * Cc + koff);
            uint4* dst = (uint4*)(Ks + nr * Cc + koff);
#pragma unroll
            for (int j = 0; j < 16; ++j) dst[j] = src[j];
        }
#endif
        // V tile transposed -> Vt[d][key] (manual; TDM cannot transpose).
        {
            int kv = tid >> 2;
            int dof = (tid & 3) * 128;
#pragma unroll
            for (int j = 0; j < 16; ++j) {
                int d0 = dof + j * 8;
                union { uint4 v; unsigned short s[8]; } wv;
                wv.v = *(const uint4*)(V + (size_t)(kb + kv) * Cc + d0);
#pragma unroll
                for (int i = 0; i < 8; ++i) Vt[(d0 + i) * 64 + kv] = wv.s[i];
            }
        }
#if HAVE_TDM
        if (wave == 0) {
            if (kt + 1 < kTiles) {
                unsigned short* Kn = (kt & 1) ? K0 : K1;
                tdm_copy_2b((unsigned int)(size_t)(void*)Kn,
                            K + (size_t)(kb + 64) * Cc, 64 * Cc);
                __builtin_amdgcn_s_wait_tensorcnt(1);  // current tile done
            } else {
                __builtin_amdgcn_s_wait_tensorcnt(0);
            }
        }
#endif
        __syncthreads();

        // S = Q * K^T (scale folded into Q)
        {
            v8f s = {0.f, 0.f, 0.f, 0.f, 0.f, 0.f, 0.f, 0.f};
            for (int ks = 0; ks < 16; ++ks) {
                v16bf a = load_frag_ab(Qs, smi * 16, Cc, ks * 32, lane);
                v16bf b = load_frag_ab(Ks, snj * 16, Cc, ks * 32, lane);
                s = wmma_bf16(a, b, s);
            }
            const int nc = lane & 15, half = lane >> 4;
#pragma unroll
            for (int r = 0; r < 8; ++r)
                Ss[(smi * 16 + r + half * 8) * 64 + snj * 16 + nc] = s[r];
        }
        __syncthreads();

        // Online softmax (one thread per query row)
        if (tid < 32) {
            const int row = tid;
            float tm = -3.0e38f;
            for (int j = 0; j < 64; ++j) tm = fmaxf(tm, Ss[row * 64 + j]);
            float mold = rmax[row];
            float mnew = fmaxf(mold, tm);
            float alpha = __expf(mold - mnew);
            float psum = 0.f;
            for (int j = 0; j < 64; ++j) {
                float pv = __expf(Ss[row * 64 + j] - mnew);
                Ps[row * 64 + j] = f2bf(pv);
                psum += pv;
            }
            rsum[row] = rsum[row] * alpha + psum;
            rmax[row] = mnew;
            rscale[row] = alpha;
        }
        __syncthreads();

        // Rescale running O accumulator
        for (int i = tid; i < 32 * 512; i += 256)
            Oacc[i] *= rscale[i >> 9];
        __syncthreads();

        // O += P @ V_tile  (M=32, K=64, N=512)
        {
            const int omi = wave & 1;
            v16bf a0 = load_frag_ab(Ps, omi * 16, 64, 0, lane);
            v16bf a1 = load_frag_ab(Ps, omi * 16, 64, 32, lane);
#pragma unroll
            for (int j = 0; j < 8; ++j) {
                int nt = (wave >> 1) + j * 4;
                v16bf b0 = load_frag_ab(Vt, nt * 16, 64, 0, lane);
                v16bf b1 = load_frag_ab(Vt, nt * 16, 64, 32, lane);
                v8f c = load_frag_c(Oacc, omi * 16, Cc, nt * 16, lane);
                c = wmma_bf16(a0, b0, c);
                c = wmma_bf16(a1, b1, c);
                store_frag_c(Oacc, omi * 16, Cc, nt * 16, lane, c);
            }
        }
        __syncthreads();
    }

    // Finalize
    for (int i = tid; i < 32 * 512; i += 256) {
        int m = i >> 9;
        int d = i & 511;
        float v = Oacc[i] / rsum[m];
        O[(size_t)(qbase + m) * Cc + d] = f2bf(v);
    }
}

// ---------------------------------------------------------------------------
// Host orchestration
// ---------------------------------------------------------------------------
extern "C" void kernel_launch(void* const* d_in, const int* in_sizes, int n_in,
                              void* d_out, int out_size, void* d_ws, size_t ws_size,
                              hipStream_t stream) {
    (void)in_sizes; (void)n_in; (void)out_size; (void)ws_size;

    const float* X = (const float*)d_in[0];
    const float* r0n1g = (const float*)d_in[1];
    const float* r0n1b = (const float*)d_in[2];
    const float* r0c1w = (const float*)d_in[3];
    const float* r0c1b = (const float*)d_in[4];
    const float* r0n2g = (const float*)d_in[5];
    const float* r0n2b = (const float*)d_in[6];
    const float* r0c2w = (const float*)d_in[7];
    const float* r0c2b = (const float*)d_in[8];
    const float* r1n1g = (const float*)d_in[9];
    const float* r1n1b = (const float*)d_in[10];
    const float* r1c1w = (const float*)d_in[11];
    const float* r1c1b = (const float*)d_in[12];
    const float* r1n2g = (const float*)d_in[13];
    const float* r1n2b = (const float*)d_in[14];
    const float* r1c2w = (const float*)d_in[15];
    const float* r1c2b = (const float*)d_in[16];
    const float* agng = (const float*)d_in[17];
    const float* agnb = (const float*)d_in[18];
    const float* aqw = (const float*)d_in[19];
    const float* aqb = (const float*)d_in[20];
    const float* akw = (const float*)d_in[21];
    const float* akb = (const float*)d_in[22];
    const float* avw = (const float*)d_in[23];
    const float* avb = (const float*)d_in[24];
    const float* aow = (const float*)d_in[25];
    const float* aob = (const float*)d_in[26];

    // Workspace carving (~113 MB)
    float* t0 = (float*)d_ws;
    float* x1 = t0 + SZ;
    float* x2 = x1 + SZ;
    unsigned short* g16   = (unsigned short*)(x2 + SZ);
    unsigned short* wconv = g16 + SZ;
    unsigned short* wproj = wconv + (size_t)Cc * CONVK;
    unsigned short* q16   = wproj + (size_t)Cc * Cc;
    unsigned short* k16   = q16 + SZ;
    unsigned short* v16b  = k16 + SZ;
    unsigned short* o16   = v16b + SZ;

    const dim3 convGrid(Ll / 64, Cc / 64);
    const dim3 gemmGrid(Ll / 64, Cc / 64);
    const int attnSmem = (32 * 512 + 32 * 64 + 96) * 4 +
                         (32 * 512 + 2 * 64 * 512 + 512 * 64 + 32 * 64) * 2;

    auto cvt = [&](const float* src, unsigned short* dst, int n) {
        f32_to_bf16_kernel<<<512, 256, 0, stream>>>(src, dst, n);
    };
    auto cvtw = [&](const float* src, unsigned short* dst) {
        convw_permute_kernel<<<512, 256, 0, stream>>>(src, dst);
    };

    // ---- ResNet block 0 ----
    gn_kernel<<<32, 256, 0, stream>>>(X, r0n1g, r0n1b, nullptr, g16, 1, 0);
    cvtw(r0c1w, wconv);
    conv3d_igemm_kernel<<<convGrid, 256, 0, stream>>>(wconv, g16, r0c1b, nullptr, t0);
    gn_kernel<<<32, 256, 0, stream>>>(t0, r0n2g, r0n2b, nullptr, g16, 1, 0);
    cvtw(r0c2w, wconv);
    conv3d_igemm_kernel<<<convGrid, 256, 0, stream>>>(wconv, g16, r0c2b, X, x1);

    // ---- Attention ----
    gn_kernel<<<32, 256, 0, stream>>>(x1, agng, agnb, nullptr, g16, 0, 1);
    cvt(aqw, wproj, Cc * Cc);
    gemm_proj_kernel<<<gemmGrid, 256, 0, stream>>>(g16, wproj, aqb, QSCALE, 0,
                                                   nullptr, nullptr, q16);
    cvt(akw, wproj, Cc * Cc);
    gemm_proj_kernel<<<gemmGrid, 256, 0, stream>>>(g16, wproj, akb, 1.0f, 0,
                                                   nullptr, nullptr, k16);
    cvt(avw, wproj, Cc * Cc);
    gemm_proj_kernel<<<gemmGrid, 256, 0, stream>>>(g16, wproj, avb, 1.0f, 0,
                                                   nullptr, nullptr, v16b);
    attention_kernel<<<Ll / 32, 256, attnSmem, stream>>>(q16, k16, v16b, o16);
    cvt(aow, wproj, Cc * Cc);
    gemm_proj_kernel<<<gemmGrid, 256, 0, stream>>>(o16, wproj, aob, 1.0f, 1,
                                                   x1, x2, nullptr);

    // ---- ResNet block 1 ----
    gn_kernel<<<32, 256, 0, stream>>>(x2, r1n1g, r1n1b, nullptr, g16, 1, 0);
    cvtw(r1c1w, wconv);
    conv3d_igemm_kernel<<<convGrid, 256, 0, stream>>>(wconv, g16, r1c1b, nullptr, t0);
    gn_kernel<<<32, 256, 0, stream>>>(t0, r1n2g, r1n2b, nullptr, g16, 1, 0);
    cvtw(r1c2w, wconv);
    conv3d_igemm_kernel<<<convGrid, 256, 0, stream>>>(wconv, g16, r1c2b, x2,
                                                      (float*)d_out);
}
